// GCNLayer_22385369547440
// MI455X (gfx1250) — matrix operand order repaired and checked by reading, not compiled, not run on hardware
//
#include <hip/hip_runtime.h>

// Problem constants (from reference): N=8192 nodes, E=262144 edges, D=512.
#define NN 8192
#define EE 262144
#define DD 512

typedef __attribute__((ext_vector_type(16))) _Float16 v16h;
typedef __attribute__((ext_vector_type(8)))  _Float16 v8h;
typedef __attribute__((ext_vector_type(8)))  float    v8f;

// ---------------- workspace layout (all offsets 256B-aligned) ----------------
static constexpr size_t MASK_WORDS = (size_t)NN * NN / 32;          // 2,097,152 u32
static constexpr size_t OFF_MASK = 0;                                // 8 MiB bitmask
static constexpr size_t OFF_CNT  = OFF_MASK + (size_t)NN * NN / 8;   // unique-edge counter
static constexpr size_t OFF_DEGC = OFF_CNT  + 256;                   // u32[NN] degree counts
static constexpr size_t OFF_DINV = OFF_DEGC + (size_t)NN * 4;        // f32[NN] rsqrt(degree)
static constexpr size_t OFF_UI   = OFF_DINV + (size_t)NN * 4;        // u32[EE] unique src
static constexpr size_t OFF_UJ   = OFF_UI   + (size_t)EE * 4;        // u32[EE] unique dst
static constexpr size_t OFF_XH   = OFF_UJ   + (size_t)EE * 4;        // f16[NN*DD] X in half
static constexpr size_t OFF_WT   = OFF_XH   + (size_t)NN * DD * 2;   // f16[DD*DD] W^T in half
static constexpr size_t OFF_SUP  = OFF_WT   + (size_t)DD * DD * 2;   // f32[NN*DD] support = XW

// ---------------- kernels ----------------

__global__ void k_init_ws(unsigned int* __restrict__ mask,
                          unsigned int* __restrict__ degc,
                          unsigned int* __restrict__ cnt) {
    size_t t = (size_t)blockIdx.x * blockDim.x + threadIdx.x;
    if (t < MASK_WORDS) mask[t] = 0u;
    if (t < NN)         degc[t] = 0u;
    if (t == 0)         *cnt = 0u;
}

__global__ void k_cvt_x(const float* __restrict__ x, _Float16* __restrict__ xh) {
    int t = blockIdx.x * blockDim.x + threadIdx.x;
    xh[t] = (_Float16)x[t];
}

// W is [DD][DD] row-major; store W^T in half so B-fragment K is contiguous per lane.
__global__ void k_cvt_wT(const float* __restrict__ w, _Float16* __restrict__ wt) {
    int t = blockIdx.x * blockDim.x + threadIdx.x;   // t = k*DD + n
    int k = t >> 9;
    int n = t & (DD - 1);
    wt[(size_t)n * DD + k] = (_Float16)w[t];
}

// Scatter-set dedup: bit (i*NN+j) in an 8MB bitmask. First setter appends the
// edge to a compacted unique list and bumps degree(i) (row-sum of S).
__global__ void k_dedup(const long long* __restrict__ ei,
                        unsigned int* __restrict__ mask,
                        unsigned int* __restrict__ cnt,
                        unsigned int* __restrict__ degc,
                        unsigned int* __restrict__ uI,
                        unsigned int* __restrict__ uJ) {
    int e = blockIdx.x * blockDim.x + threadIdx.x;
    if (e >= EE) return;
    unsigned int i = (unsigned int)ei[e];
    unsigned int j = (unsigned int)ei[EE + e];
    unsigned int b = i * (unsigned int)NN + j;       // < 2^26
    unsigned int m = 1u << (b & 31u);
    unsigned int old = atomicOr(&mask[b >> 5], m);
    if ((old & m) == 0u) {
        unsigned int pos = atomicAdd(cnt, 1u);
        uI[pos] = i;
        uJ[pos] = j;
        atomicAdd(&degc[i], 1u);
    }
}

__global__ void k_dinv(const unsigned int* __restrict__ degc, float* __restrict__ dinv) {
    int i = blockIdx.x * blockDim.x + threadIdx.x;
    dinv[i] = rsqrtf((float)degc[i] + 1.0f);         // +1 = self loop from identity
}

// A 16x32 f16 fragment: lane m holds contiguous K = 16*half .. 16*half+15
__device__ __forceinline__ v16h load_fragA(const _Float16* p) {
    v8h a0 = *(const v8h*)(p);
    v8h a1 = *(const v8h*)(p + 8);
    v16h a;
    #pragma unroll
    for (int q = 0; q < 8; ++q) { a[q] = a0[q]; a[8 + q] = a1[q]; }
    return a;
}

// B 32x16 f16 fragment: lane n holds contiguous K = 16*half .. 16*half+15
__device__ __forceinline__ v16h load_fragB(const _Float16* p) {
    v8h b0 = *(const v8h*)(p);
    v8h b1 = *(const v8h*)(p + 8);
    v16h b;
    #pragma unroll
    for (int q = 0; q < 8; ++q) { b[q] = b0[q]; b[8 + q] = b1[q]; }
    return b;
}

// support = X @ W via v_wmma_f32_16x16x32_f16.
// One wave owns one 16-row M-tile and four 16-col N-tiles (A fragment reused 4x).
// Register double-buffered k-loop: next iteration's fragment loads are issued
// before the current iteration's 4 WMMAs, so the L2 round-trip overlaps compute
// instead of hitting a full s_wait_loadcnt 0 stall in front of every WMMA.
__global__ void __launch_bounds__(128) k_gemm(const _Float16* __restrict__ Xh,
                                              const _Float16* __restrict__ Wt,
                                              float* __restrict__ sup) {
    const int lane  = threadIdx.x & 31;
    const int wave  = blockIdx.x * (blockDim.x >> 5) + (threadIdx.x >> 5);
    const int mTile = wave >> 3;       // 0..511  (8192/16)
    const int nGrp  = wave & 7;        // 0..7    (512/64)
    const int m     = lane & 15;
    const int half  = lane >> 4;
    const int nBase = nGrp * 64;

    const _Float16* pa = Xh + (size_t)(mTile * 16 + m) * DD + half * 16;
    const _Float16* pb0 = Wt + (size_t)(nBase +  0 + m) * DD + half * 16;
    const _Float16* pb1 = Wt + (size_t)(nBase + 16 + m) * DD + half * 16;
    const _Float16* pb2 = Wt + (size_t)(nBase + 32 + m) * DD + half * 16;
    const _Float16* pb3 = Wt + (size_t)(nBase + 48 + m) * DD + half * 16;

    v8f acc[4] = {v8f{}, v8f{}, v8f{}, v8f{}};

    // prologue: fragments for kk = 0
    v16h a_cur = load_fragA(pa);
    v16h b_cur[4];
    b_cur[0] = load_fragB(pb0);
    b_cur[1] = load_fragB(pb1);
    b_cur[2] = load_fragB(pb2);
    b_cur[3] = load_fragB(pb3);

    #pragma unroll 2
    for (int kk = 32; kk < DD; kk += 32) {
        // issue next iteration's loads first (overlap with the WMMA burst)
        v16h a_nxt = load_fragA(pa + kk);
        v16h b_nxt[4];
        b_nxt[0] = load_fragB(pb0 + kk);
        b_nxt[1] = load_fragB(pb1 + kk);
        b_nxt[2] = load_fragB(pb2 + kk);
        b_nxt[3] = load_fragB(pb3 + kk);

        #pragma unroll
        for (int t = 0; t < 4; ++t) {
            acc[t] = __builtin_amdgcn_wmma_f32_16x16x32_f16(
                false, a_cur, false, b_cur[t], (short)0, acc[t], false, false);
        }
        a_cur = a_nxt;
        #pragma unroll
        for (int t = 0; t < 4; ++t) b_cur[t] = b_nxt[t];
    }

    // epilogue: last k-step
    #pragma unroll
    for (int t = 0; t < 4; ++t) {
        acc[t] = __builtin_amdgcn_wmma_f32_16x16x32_f16(
            false, a_cur, false, b_cur[t], (short)0, acc[t], false, false);
    }

    // D 16x16 f32: VGPR r -> row = r + 8*half, col = lane&15
    #pragma unroll
    for (int t = 0; t < 4; ++t) {
        #pragma unroll
        for (int r = 0; r < 8; ++r) {
            int row = mTile * 16 + r + 8 * half;
            int col = nBase + t * 16 + m;
            sup[(size_t)row * DD + col] = acc[t][r];
        }
    }
}

// out = d_i^2 * support[i] + bias  (the identity self-loop term)
__global__ void k_out_init(const float* __restrict__ sup,
                           const float* __restrict__ dinv,
                           const float* __restrict__ bias,
                           float* __restrict__ out) {
    int t = blockIdx.x * blockDim.x + threadIdx.x;
    int i = t >> 9;
    int c = t & (DD - 1);
    float di = dinv[i];
    out[t] = di * di * sup[t] + bias[c];
}

// out[i,:] += d_i * d_j * support[j,:] for every unique edge. One 128-thread
// block per edge; 4 columns per thread; out/support are L2-resident (16MB each).
__global__ void __launch_bounds__(128) k_scatter(const unsigned int* __restrict__ uI,
                                                 const unsigned int* __restrict__ uJ,
                                                 const unsigned int* __restrict__ cnt,
                                                 const float* __restrict__ dinv,
                                                 const float* __restrict__ sup,
                                                 float* __restrict__ out) {
    unsigned int u = blockIdx.x;
    if (u >= *cnt) return;
    unsigned int i = uI[u];
    unsigned int j = uJ[u];
    float coef = dinv[i] * dinv[j];
    const float4* s4 = (const float4*)(sup + (size_t)j * DD);
    float* o = out + (size_t)i * DD + threadIdx.x * 4;
    float4 v = s4[threadIdx.x];
    atomicAdd(&o[0], coef * v.x);
    atomicAdd(&o[1], coef * v.y);
    atomicAdd(&o[2], coef * v.z);
    atomicAdd(&o[3], coef * v.w);
}

// ---------------- host launch ----------------
extern "C" void kernel_launch(void* const* d_in, const int* in_sizes, int n_in,
                              void* d_out, int out_size, void* d_ws, size_t ws_size,
                              hipStream_t stream) {
    const float*     x    = (const float*)d_in[0];       // [NN, DD] f32
    const long long* ei   = (const long long*)d_in[1];   // [2, EE] i64
    const float*     w    = (const float*)d_in[2];       // [DD, DD] f32
    const float*     bias = (const float*)d_in[3];       // [DD] f32
    float*           out  = (float*)d_out;               // [NN, DD] f32

    char* ws = (char*)d_ws;
    unsigned int* mask = (unsigned int*)(ws + OFF_MASK);
    unsigned int* cnt  = (unsigned int*)(ws + OFF_CNT);
    unsigned int* degc = (unsigned int*)(ws + OFF_DEGC);
    float*        dinv = (float*)(ws + OFF_DINV);
    unsigned int* uI   = (unsigned int*)(ws + OFF_UI);
    unsigned int* uJ   = (unsigned int*)(ws + OFF_UJ);
    _Float16*     Xh   = (_Float16*)(ws + OFF_XH);
    _Float16*     Wt   = (_Float16*)(ws + OFF_WT);
    float*        sup  = (float*)(ws + OFF_SUP);

    // 1. zero bitmask / counters (every call: determinism)
    k_init_ws<<<(unsigned)(MASK_WORDS / 256), 256, 0, stream>>>(mask, degc, cnt);
    // 2. precision convert
    k_cvt_x <<<(NN * DD) / 256, 256, 0, stream>>>(x, Xh);
    k_cvt_wT<<<(DD * DD) / 256, 256, 0, stream>>>(w, Wt);
    // 3. dedup + degree
    k_dedup <<<EE / 256, 256, 0, stream>>>(ei, mask, cnt, degc, uI, uJ);
    // 4. D^{-1/2}
    k_dinv  <<<NN / 256, 256, 0, stream>>>(degc, dinv);
    // 5. support = X @ W (WMMA): 4096 waves, 4 waves/block, double-buffered k-loop
    k_gemm  <<<1024, 128, 0, stream>>>(Xh, Wt, sup);
    // 6. self-loop term + bias
    k_out_init<<<(NN * DD) / 256, 256, 0, stream>>>(sup, dinv, bias, out);
    // 7. sparse aggregation (one block per potential unique edge)
    k_scatter<<<EE, 128, 0, stream>>>(uI, uJ, cnt, dinv, sup, out);
}
